// GlossAttention_88476326297951
// MI455X (gfx1250) — matrix operand order repaired and checked by reading, not compiled, not run on hardware
//
#include <hip/hip_runtime.h>

// ---------------------------------------------------------------------------
// GlossAttention for MI455X (gfx1250, wave32, WMMA).
// All matmuls via v_wmma_f32_16x16x32_bf16 (f32 accumulate); softmax f32 VALU.
// ---------------------------------------------------------------------------

typedef __attribute__((ext_vector_type(16))) __bf16 v16bf;
typedef __attribute__((ext_vector_type(8)))  __bf16 v8bf;
typedef __attribute__((ext_vector_type(8)))  float  v8f;

union BF16x16 { v16bf v; v8bf h[2]; };

#define C_DIM   1024
#define HEADS   16
#define HDIM    64
#define BATCH   4
#define NX      2048
#define NQ      512
#define SCALE_F 0.125f   // 64^-0.5 (folded into Q projection)

#define MODE_KV   0
#define MODE_Q    1
#define MODE_PROJ 2

// ---------------------------------------------------------------------------
// Tiled GEMM:  C(MxNc) = A(MxK) @ Bw(KxNc); A f32 (bf16 for MODE_PROJ),
// Bw f32.  Compile-time MODE -> straight-line staging, no runtime branches.
// Block = 128 threads (4 waves); block tile 64x64: wave w owns rows
// [w*16, w*16+16) x all 64 cols (4 WMMA col-tiles, A-fragment reused 4x).
// ---------------------------------------------------------------------------
template <int MODE>
__global__ __launch_bounds__(128)
void gemm_wmma_kernel(const void* __restrict__ Ap, const float* __restrict__ Bw,
                      int K, int Nc,
                      __bf16* __restrict__ kbf, __bf16* __restrict__ vt,
                      __bf16* __restrict__ qbf,
                      float* __restrict__ outp, const float* __restrict__ bias)
{
    __shared__ __align__(64) __bf16 Atile[64][32];   // 4 KB, [m][k]
    __shared__ __align__(64) __bf16 Btile[64][32];   // 4 KB, transposed [n][k]

    const int tid  = threadIdx.x;
    const int wave = tid >> 5;
    const int lane = tid & 31;
    const int half = (lane >> 4) & 1;
    const int l15  = lane & 15;
    const int mbase = blockIdx.x * 64;
    const int nbase = blockIdx.y * 64;

    const float*  Af = (const float*)Ap;
    const __bf16* Ab = (const __bf16*)Ap;

    v8f acc[4] = {{}, {}, {}, {}};

    for (int k0 = 0; k0 < K; k0 += 32) {
        // ---- stage A 64x32: thread -> 16 contiguous floats of one row ----
        {
            const int r  = tid >> 1;
            const int c0 = (tid & 1) * 16;
            BF16x16 u;
            if (MODE == MODE_PROJ) {
                // A already bf16: one 32B vector load
                u.v = *reinterpret_cast<const v16bf*>(
                          Ab + (size_t)(mbase + r) * K + k0 + c0);
            } else {
                const float* src = Af + (size_t)(mbase + r) * K + k0 + c0;
                float tmp[16];
                *(float4*)&tmp[0]  = *reinterpret_cast<const float4*>(src);
                *(float4*)&tmp[4]  = *reinterpret_cast<const float4*>(src + 4);
                *(float4*)&tmp[8]  = *reinterpret_cast<const float4*>(src + 8);
                *(float4*)&tmp[12] = *reinterpret_cast<const float4*>(src + 12);
                #pragma unroll
                for (int j = 0; j < 16; ++j) u.v[j] = (__bf16)tmp[j];
            }
            *reinterpret_cast<v16bf*>(&Atile[r][c0]) = u.v;
        }
        // ---- stage B 32x64 -> transposed [n][k]: thread -> 16 floats of one k-row
        {
            const int kr = tid >> 2;
            const int c0 = (tid & 3) * 16;
            const float* src = Bw + (size_t)(k0 + kr) * Nc + nbase + c0;
            float tmp[16];
            *(float4*)&tmp[0]  = *reinterpret_cast<const float4*>(src);
            *(float4*)&tmp[4]  = *reinterpret_cast<const float4*>(src + 4);
            *(float4*)&tmp[8]  = *reinterpret_cast<const float4*>(src + 8);
            *(float4*)&tmp[12] = *reinterpret_cast<const float4*>(src + 12);
            #pragma unroll
            for (int j = 0; j < 16; ++j) Btile[c0 + j][kr] = (__bf16)tmp[j];
        }
        __syncthreads();

        // A fragment: row = l15 (+wave strip),
        // K elem j -> half*16 + j (contig 32B, matches B)
        BF16x16 afr;
        afr.v = *reinterpret_cast<const v16bf*>(&Atile[wave * 16 + l15][half * 16]);

        #pragma unroll
        for (int ct = 0; ct < 4; ++ct) {
            // B fragment: col = ct*16 + l15, K elem j -> half*16 + j (contig 32B)
            v16bf bfr = *reinterpret_cast<const v16bf*>(&Btile[ct * 16 + l15][half * 16]);
            acc[ct] = __builtin_amdgcn_wmma_f32_16x16x32_bf16(
                          false, afr.v, false, bfr, (short)0, acc[ct], false, false);
        }
        __syncthreads();
    }

    // C/D layout: VGPR r -> row (r + 8*half), col l15
    #pragma unroll
    for (int ct = 0; ct < 4; ++ct) {
        #pragma unroll
        for (int r = 0; r < 8; ++r) {
            const int grow = mbase + wave * 16 + r + 8 * half;
            const int gcol = nbase + ct * 16 + l15;
            const float v = acc[ct][r];
            if (MODE == MODE_KV) {
                const int b = grow >> 11, n = grow & (NX - 1);   // NX = 2048
                if (gcol < C_DIM) {               // K -> [B,H,Nx,D] bf16
                    const int h = gcol >> 6, d = gcol & 63;
                    kbf[(((size_t)(b * HEADS + h)) * NX + n) * HDIM + d] = (__bf16)v;
                } else {                          // V -> transposed [B,H,D,Nx] bf16
                    const int c2 = gcol - C_DIM;
                    const int h = c2 >> 6, d = c2 & 63;
                    vt[(((size_t)(b * HEADS + h)) * HDIM + d) * NX + n] = (__bf16)v;
                }
            } else if (MODE == MODE_Q) {          // Q*SCALE -> [B,H,N,D] bf16
                const int b = grow >> 9, n = grow & (NQ - 1);    // NQ = 512
                const int h = gcol >> 6, d = gcol & 63;
                qbf[(((size_t)(b * HEADS + h)) * NQ + n) * HDIM + d]
                    = (__bf16)(v * SCALE_F);
            } else {                              // final projection + bias, f32 out
                outp[(size_t)grow * Nc + gcol] = v + bias[gcol];
            }
        }
    }
}

// ---------------------------------------------------------------------------
// Gate kernel: g = x @ w_la  -> per (b,h): 0.5*mean_n(g) + 0.5*max_n(g).
// ---------------------------------------------------------------------------
__global__ __launch_bounds__(256)
void gate_kernel(const float* __restrict__ x, const float* __restrict__ wla,
                 float* __restrict__ gate)
{
    __shared__ float wlaS[C_DIM][HEADS];   // 64 KB
    __shared__ float red[256][HEADS];      // 16 KB
    const int b = blockIdx.x, tid = threadIdx.x;

    for (int i = tid; i < C_DIM * HEADS; i += 256)
        wlaS[i >> 4][i & 15] = wla[i];
    __syncthreads();

    float lsum[HEADS], lmax[HEADS];
    #pragma unroll
    for (int h = 0; h < HEADS; ++h) { lsum[h] = 0.f; lmax[h] = -3.4e38f; }

    for (int n = tid; n < NX; n += 256) {
        const float* xr = x + ((size_t)b * NX + n) * C_DIM;
        float acc[HEADS];
        #pragma unroll
        for (int h = 0; h < HEADS; ++h) acc[h] = 0.f;
        for (int c = 0; c < C_DIM; ++c) {
            const float xv = xr[c];
            #pragma unroll
            for (int h = 0; h < HEADS; ++h) acc[h] = fmaf(xv, wlaS[c][h], acc[h]);
        }
        #pragma unroll
        for (int h = 0; h < HEADS; ++h) {
            lsum[h] += acc[h];
            lmax[h] = fmaxf(lmax[h], acc[h]);
        }
    }

    #pragma unroll
    for (int h = 0; h < HEADS; ++h) red[tid][h] = lsum[h];
    __syncthreads();
    if (tid < HEADS) {
        float s = 0.f;
        for (int t = 0; t < 256; ++t) s += red[t][tid];
        gate[b * HEADS + tid] = s;   // stash sum temporarily
    }
    __syncthreads();
    #pragma unroll
    for (int h = 0; h < HEADS; ++h) red[tid][h] = lmax[h];
    __syncthreads();
    if (tid < HEADS) {
        float m = -3.4e38f;
        for (int t = 0; t < 256; ++t) m = fmaxf(m, red[t][tid]);
        const float mean = gate[b * HEADS + tid] * (1.0f / (float)NX);
        gate[b * HEADS + tid] = 0.5f * mean + 0.5f * m;
    }
}

// ---------------------------------------------------------------------------
// Attention kernel: one workgroup (4 waves) per (b,h, 16-query tile).
//   Phase 1: S = (Q*SCALE) Kt  (bf16 WMMA), S kept in 128 KB LDS (f32).
//   Phase 2: softmax -> a;  stats of gate*a;  P = w*softmax(gate*a)+(1-w)*a.
//   Phase 3: out = P @ V    (bf16 WMMA, one d-tile per wave) -> ctx bf16.
// ---------------------------------------------------------------------------
__global__ __launch_bounds__(128)
void attn_kernel(const __bf16* __restrict__ qbf, const __bf16* __restrict__ kbf,
                 const __bf16* __restrict__ vt, const float* __restrict__ gate,
                 const float* __restrict__ attw, __bf16* __restrict__ ctx)
{
    __shared__ __align__(64) float S[16][NX];          // 128 KB
    __shared__ float stats[16][8];
    __shared__ float rowmax[16], rowsum[16], rowmax2[16], rowsum2[16];

    const int tid  = threadIdx.x;
    const int wave = tid >> 5;
    const int lane = tid & 31;
    const int half = (lane >> 4) & 1;
    const int l15  = lane & 15;

    const int ntiles = NQ / 16;                 // 32
    const int nt = blockIdx.x % ntiles;
    const int bh = blockIdx.x / ntiles;         // b*HEADS + h
    const size_t kbase = (size_t)bh * NX * HDIM;
    const size_t qrow  = ((size_t)bh * NQ + nt * 16 + l15) * HDIM;

    // Q A-fragments for d=[0,32) and d=[32,64)
    BF16x16 qa[2];
    #pragma unroll
    for (int f = 0; f < 2; ++f) {
        qa[f].v = *reinterpret_cast<const v16bf*>(qbf + qrow + f * 32 + half * 16);
    }

    // -------- Phase 1: scores (scale already folded into Q) --------
    for (int mt = wave; mt < NX / 16; mt += 4) {
        const size_t krow = kbase + (size_t)(mt * 16 + l15) * HDIM;
        v8f acc = {};
        #pragma unroll
        for (int f = 0; f < 2; ++f) {
            // B elem j: d = f*32 + half*16 + j, col m = mt*16 + l15 (contig 32B)
            v16bf bk = *reinterpret_cast<const v16bf*>(kbf + krow + f * 32 + half * 16);
            acc = __builtin_amdgcn_wmma_f32_16x16x32_bf16(
                      false, qa[f].v, false, bk, (short)0, acc, false, false);
        }
        #pragma unroll
        for (int r = 0; r < 8; ++r)
            S[r + 8 * half][mt * 16 + l15] = acc[r];
    }
    __syncthreads();

    // -------- Phase 2: double softmax + blend --------
    const int row = tid >> 3;        // 16 rows, 8 threads each
    const int g   = tid & 7;
    const float gt = gate[bh];
    const float wb = attw[0];

    float lm = -3.4e38f;
    for (int m = g; m < NX; m += 8) lm = fmaxf(lm, S[row][m]);
    stats[row][g] = lm;
    __syncthreads();
    if (tid < 16) {
        float m = stats[tid][0];
        for (int i = 1; i < 8; ++i) m = fmaxf(m, stats[tid][i]);
        rowmax[tid] = m;
    }
    __syncthreads();

    float rm = rowmax[row], ls = 0.f;
    for (int m = g; m < NX; m += 8) ls += __expf(S[row][m] - rm);
    stats[row][g] = ls;
    __syncthreads();
    if (tid < 16) {
        float s = 0.f;
        for (int i = 0; i < 8; ++i) s += stats[tid][i];
        rowsum[tid] = s;
    }
    __syncthreads();

    const float inv1 = 1.f / rowsum[row];
    rm = rowmax[row];
    float lm2 = -3.4e38f;
    for (int m = g; m < NX; m += 8) {
        const float a = __expf(S[row][m] - rm) * inv1;
        S[row][m] = a;                       // S now holds attn
        lm2 = fmaxf(lm2, gt * a);
    }
    stats[row][g] = lm2;
    __syncthreads();
    if (tid < 16) {
        float m = stats[tid][0];
        for (int i = 1; i < 8; ++i) m = fmaxf(m, stats[tid][i]);
        rowmax2[tid] = m;
    }
    __syncthreads();

    const float rm2 = rowmax2[row];
    float ls2 = 0.f;
    for (int m = g; m < NX; m += 8) ls2 += __expf(gt * S[row][m] - rm2);
    stats[row][g] = ls2;
    __syncthreads();
    if (tid < 16) {
        float s = 0.f;
        for (int i = 0; i < 8; ++i) s += stats[tid][i];
        rowsum2[tid] = s;
    }
    __syncthreads();

    const float inv2 = 1.f / rowsum2[row];
    for (int m = g; m < NX; m += 8) {
        const float a = S[row][m];
        S[row][m] = wb * __expf(gt * a - rm2) * inv2 + (1.f - wb) * a;  // blended P
    }
    __syncthreads();

    // -------- Phase 3: out = P @ V  (one 16-wide d-tile per wave) --------
    const int d0 = wave * 16;
    const size_t vrow = (size_t)bh * HDIM * NX + (size_t)(d0 + l15) * NX;
    v8f oacc = {};
    for (int k0 = 0; k0 < NX; k0 += 32) {
        // A fragment from P (f32 LDS -> bf16): one contiguous 16-float run
        float tmpv[16];
        const float* srow = &S[l15][0];
        *(float4*)&tmpv[0]  = *reinterpret_cast<const float4*>(srow + k0 + half * 16);
        *(float4*)&tmpv[4]  = *reinterpret_cast<const float4*>(srow + k0 + half * 16 + 4);
        *(float4*)&tmpv[8]  = *reinterpret_cast<const float4*>(srow + k0 + half * 16 + 8);
        *(float4*)&tmpv[12] = *reinterpret_cast<const float4*>(srow + k0 + half * 16 + 12);
        BF16x16 pa;
        #pragma unroll
        for (int j = 0; j < 16; ++j) pa.v[j] = (__bf16)tmpv[j];

        // B elem j: m = k0 + half*16 + j, col d = d0 + l15 (V transposed, contig)
        v16bf vb = *reinterpret_cast<const v16bf*>(vt + vrow + k0 + half * 16);

        oacc = __builtin_amdgcn_wmma_f32_16x16x32_bf16(
                   false, pa.v, false, vb, (short)0, oacc, false, false);
    }

    const int b = bh / HEADS, h = bh % HEADS;
    #pragma unroll
    for (int r = 0; r < 8; ++r) {
        const int n = nt * 16 + r + 8 * half;
        ctx[((size_t)(b * NQ + n)) * C_DIM + h * HDIM + d0 + l15] = (__bf16)oacc[r];
    }
}

// ---------------------------------------------------------------------------
extern "C" void kernel_launch(void* const* d_in, const int* in_sizes, int n_in,
                              void* d_out, int out_size, void* d_ws, size_t ws_size,
                              hipStream_t stream)
{
    (void)in_sizes; (void)n_in; (void)out_size; (void)ws_size;

    const float* x      = (const float*)d_in[0];
    const float* cls    = (const float*)d_in[1];
    const float* w_kv   = (const float*)d_in[2];
    const float* w_q    = (const float*)d_in[3];
    const float* w_la   = (const float*)d_in[4];
    const float* w_proj = (const float*)d_in[5];
    const float* b_proj = (const float*)d_in[6];
    const float* attw   = (const float*)d_in[7];
    float* out = (float*)d_out;

    // workspace carve-up (all 256B aligned)
    char* ws = (char*)d_ws;
    const size_t KV_BYTES  = (size_t)BATCH * HEADS * NX * HDIM * 2;  // 16 MB
    const size_t Q_BYTES   = (size_t)BATCH * HEADS * NQ * HDIM * 2;  //  4 MB
    const size_t CTX_BYTES = (size_t)BATCH * NQ * C_DIM * 2;         //  4 MB
    __bf16* kbf  = (__bf16*)(ws);
    __bf16* vt   = (__bf16*)(ws + KV_BYTES);
    __bf16* qbf  = (__bf16*)(ws + 2 * KV_BYTES);
    __bf16* ctx  = (__bf16*)(ws + 2 * KV_BYTES + Q_BYTES);
    float*  gate = (float*)(ws + 2 * KV_BYTES + Q_BYTES + CTX_BYTES);

    // 1) kv projection: [B*Nx, C] @ [C, 2C] -> K bf16, V^T bf16
    gemm_wmma_kernel<MODE_KV><<<dim3((BATCH * NX) / 64, (2 * C_DIM) / 64), 128, 0, stream>>>(
        x, w_kv, C_DIM, 2 * C_DIM, kbf, vt, qbf, nullptr, nullptr);

    // 2) q projection (scale folded): [B*N, C] @ [C, C] -> Q bf16
    gemm_wmma_kernel<MODE_Q><<<dim3((BATCH * NQ) / 64, C_DIM / 64), 128, 0, stream>>>(
        cls, w_q, C_DIM, C_DIM, kbf, vt, qbf, nullptr, nullptr);

    // 3) pooled gate per (b,h)
    gate_kernel<<<BATCH, 256, 0, stream>>>(x, w_la, gate);

    // 4) attention (scores + double softmax + blend + PV)
    attn_kernel<<<BATCH * HEADS * (NQ / 16), 128, 0, stream>>>(
        qbf, kbf, vt, gate, attw, ctx);

    // 5) output projection + bias -> f32 d_out
    gemm_wmma_kernel<MODE_PROJ><<<dim3((BATCH * NQ) / 64, C_DIM / 64), 128, 0, stream>>>(
        ctx, w_proj, C_DIM, C_DIM, nullptr, nullptr, nullptr, out, b_proj);
}
